// GlobalMLPBackflow_48344151884301
// MI455X (gfx1250) — matrix operand order repaired and checked by reading, not compiled, not run on hardware
//
#include <hip/hip_runtime.h>

// ---------------- CDNA5 (gfx1250) wave32 WMMA types ----------------
typedef __attribute__((ext_vector_type(16))) __bf16 v16bf;
typedef __attribute__((ext_vector_type(8)))  __bf16 v8bf;
typedef __attribute__((ext_vector_type(4)))  __bf16 v4bf;
typedef __attribute__((ext_vector_type(8)))  float  v8f;

// Problem dims (fixed by the reference)
constexpr int Bsz = 1024;
constexpr int Ssz = 36;
constexpr int Esz = 64;
constexpr int Hsz = 512;
constexpr int Psz = 128;
constexpr int Gsz = Ssz * Esz;   // 2304

// Tiling
constexpr int BM = 128;
constexpr int BN = 128;
constexpr int BK = 32;           // == WMMA K for bf16
constexpr int LDA = BK + 8;      // padded bf16 stride (80B, 16B-aligned rows)
constexpr int STG = BN + 8;      // output staging stride

union FragU { v16bf v; v8bf h[2]; };

// A-operand fragment (16x32 bf16): lane L holds row (L&15);
// lanes 0-15: K0-7 + K16-23, lanes 16-31: K8-15 + K24-31  (ISA 7.12.2)
__device__ inline v16bf load_frag_A(const __bf16* tile, int lane) {
  const __bf16* p = tile + (lane & 15) * LDA + ((lane >> 4) << 3);
  FragU f;
  f.h[0] = *(const v8bf*)(p);
  f.h[1] = *(const v8bf*)(p + 16);
  return f.v;
}

// B-operand fragment (32x16 bf16), tile stored N-major (Bt[n][k]):
// lane L holds column (L&15); lanes 0-15: K0-7 + K16-23, lanes 16-31: K8-15 + K24-31
__device__ inline v16bf load_frag_B(const __bf16* tile, int lane) {
  const __bf16* p = tile + (lane & 15) * LDA + ((lane >> 4) << 3);
  FragU f;
  f.h[0] = *(const v8bf*)(p);
  f.h[1] = *(const v8bf*)(p + 16);
  return f.v;
}

__device__ inline float gelu_exact(float v) {
  return 0.5f * v * (1.0f + erff(v * 0.70710678118654752f));
}

// =====================================================================
// Kernel 1: h[b,s,:] = gelu(x_flat[b,:] @ W1[s] + b1[s])  -> bf16 to ws
// grid = (8 Mtiles * 4 Ntiles, 36); same-s blocks adjacent for L2 reuse
// =====================================================================
__global__ __launch_bounds__(256) void k_gemm1_gelu(
    const float* __restrict__ x, const float* __restrict__ W1,
    const float* __restrict__ b1, __bf16* __restrict__ hout)
{
  __shared__ __bf16 sA[2][BM * LDA];
  __shared__ __bf16 sB[2][BN * LDA];
  __shared__ __bf16 stage[BM * STG];

  const int s    = blockIdx.y;
  const int m0   = (blockIdx.x >> 2) * BM;
  const int n0   = (blockIdx.x & 3) * BN;
  const int tid  = threadIdx.x;
  const int lane = tid & 31;
  const int wave = tid >> 5;
  const int wm   = (wave >> 1) * 32;   // 4 waves along M
  const int wn   = (wave & 1) * 64;    // 2 waves along N

  const float* Ab = x  + (size_t)m0 * Gsz;
  const float* Bb = W1 + (size_t)s * Gsz * Hsz + n0;

  // staging assignments
  const int arow = tid >> 1;           // 0..127
  const int acol = (tid & 1) * 16;     // 16 fp32 per thread (A)
  const int bkq  = (tid >> 5) * 4;     // 4x4 micro-tile (B), k base
  const int bnq  = (tid & 31) * 4;     //                    n base

  float4 ra[4], rb[4];

  auto gload = [&](int k0) {
    const float* pa = Ab + (size_t)arow * Gsz + k0 + acol;
    ra[0] = *(const float4*)(pa + 0);
    ra[1] = *(const float4*)(pa + 4);
    ra[2] = *(const float4*)(pa + 8);
    ra[3] = *(const float4*)(pa + 12);
    const float* pb = Bb + (size_t)(k0 + bkq) * Hsz + bnq;
    rb[0] = *(const float4*)(pb);
    rb[1] = *(const float4*)(pb + Hsz);
    rb[2] = *(const float4*)(pb + 2 * Hsz);
    rb[3] = *(const float4*)(pb + 3 * Hsz);
  };

  auto lstore = [&](int buf) {
    // A: row-major [BM][LDA], fp32 -> bf16
    v8bf p0, p1;
    const float* fa = (const float*)ra;
#pragma unroll
    for (int i = 0; i < 8; ++i) { p0[i] = (__bf16)fa[i]; p1[i] = (__bf16)fa[8 + i]; }
    *(v8bf*)&sA[buf][arow * LDA + acol]     = p0;
    *(v8bf*)&sA[buf][arow * LDA + acol + 8] = p1;
    // B: transpose 4x4 micro-tile into N-major [BN][LDA]
#pragma unroll
    for (int j = 0; j < 4; ++j) {
      v4bf t;
#pragma unroll
      for (int i = 0; i < 4; ++i) t[i] = (__bf16)(((const float*)&rb[i])[j]);
      *(v4bf*)&sB[buf][(bnq + j) * LDA + bkq] = t;
    }
  };

  v8f acc[2][4];
#pragma unroll
  for (int fm = 0; fm < 2; ++fm)
#pragma unroll
    for (int fn = 0; fn < 4; ++fn)
#pragma unroll
      for (int i = 0; i < 8; ++i) acc[fm][fn][i] = 0.0f;

  constexpr int KT = Gsz / BK;   // 72
  gload(0);
  lstore(0);
  __syncthreads();

  for (int t = 0; t < KT; ++t) {
    const int cur = t & 1;
    if (t + 1 < KT) gload((t + 1) * BK);
    if (t + 2 < KT)   // pull K+2 slab of W1 toward L2 (global_prefetch_b8)
      __builtin_prefetch(Bb + (size_t)((t + 2) * BK + bkq) * Hsz + bnq, 0, 1);

    v16bf af[2], bf_[4];
#pragma unroll
    for (int fm = 0; fm < 2; ++fm)
      af[fm] = load_frag_A(&sA[cur][(wm + fm * 16) * LDA], lane);
#pragma unroll
    for (int fn = 0; fn < 4; ++fn)
      bf_[fn] = load_frag_B(&sB[cur][(wn + fn * 16) * LDA], lane);

#pragma unroll
    for (int fm = 0; fm < 2; ++fm)
#pragma unroll
      for (int fn = 0; fn < 4; ++fn)
        acc[fm][fn] = __builtin_amdgcn_wmma_f32_16x16x32_bf16(
            false, af[fm], false, bf_[fn], (short)0, acc[fm][fn], false, false);

    if (t + 1 < KT) lstore(1 - cur);
    __syncthreads();
  }

  // bias + exact GELU, stage as bf16 for coalesced store of h
  float bias[4];
#pragma unroll
  for (int fn = 0; fn < 4; ++fn)
    bias[fn] = b1[s * Hsz + n0 + wn + fn * 16 + (lane & 15)];

  const int mro = (lane >> 4) * 8;     // C/D layout: VGPR r -> M = r (+8 for hi lanes)
#pragma unroll
  for (int fm = 0; fm < 2; ++fm)
#pragma unroll
    for (int fn = 0; fn < 4; ++fn)
#pragma unroll
      for (int r = 0; r < 8; ++r) {
        float v = acc[fm][fn][r] + bias[fn];
        v = gelu_exact(v);
        stage[(wm + fm * 16 + mro + r) * STG + wn + fn * 16 + (lane & 15)] = (__bf16)v;
      }
  __syncthreads();

  const int srow = tid >> 1;
  const int scol = (tid & 1) * 64;
  __bf16* dst = hout + (size_t)(m0 + srow) * (Ssz * Hsz) + (size_t)s * Hsz + n0 + scol;
#pragma unroll
  for (int j = 0; j < 8; ++j)
    *(v8bf*)(dst + j * 8) = *(const v8bf*)&stage[srow * STG + scol + j * 8];
}

// =====================================================================
// Kernel 2: out[b,s,:] = h[b,s,:] @ W2[s] + b2[s]   (BN = P = 128)
// grid = (8, 36)
// =====================================================================
__global__ __launch_bounds__(256) void k_gemm2(
    const __bf16* __restrict__ h, const float* __restrict__ W2,
    const float* __restrict__ b2, float* __restrict__ out)
{
  __shared__ __bf16 sA[2][BM * LDA];
  __shared__ __bf16 sB[2][BN * LDA];

  const int s    = blockIdx.y;
  const int m0   = blockIdx.x * BM;
  const int tid  = threadIdx.x;
  const int lane = tid & 31;
  const int wave = tid >> 5;
  const int wm   = (wave >> 1) * 32;
  const int wn   = (wave & 1) * 64;

  const __bf16* Ab = h  + (size_t)m0 * (Ssz * Hsz) + (size_t)s * Hsz;
  const float*  Bb = W2 + (size_t)s * Hsz * Psz;

  const int arow = tid >> 1;
  const int acol = (tid & 1) * 16;
  const int bkq  = (tid >> 5) * 4;
  const int bnq  = (tid & 31) * 4;

  v8bf la0, la1;
  float4 rb[4];

  auto gload = [&](int k0) {
    const __bf16* pa = Ab + (size_t)arow * (Ssz * Hsz) + k0 + acol;
    la0 = *(const v8bf*)(pa);
    la1 = *(const v8bf*)(pa + 8);
    const float* pb = Bb + (size_t)(k0 + bkq) * Psz + bnq;
    rb[0] = *(const float4*)(pb);
    rb[1] = *(const float4*)(pb + Psz);
    rb[2] = *(const float4*)(pb + 2 * Psz);
    rb[3] = *(const float4*)(pb + 3 * Psz);
  };

  auto lstore = [&](int buf) {
    *(v8bf*)&sA[buf][arow * LDA + acol]     = la0;
    *(v8bf*)&sA[buf][arow * LDA + acol + 8] = la1;
#pragma unroll
    for (int j = 0; j < 4; ++j) {
      v4bf t;
#pragma unroll
      for (int i = 0; i < 4; ++i) t[i] = (__bf16)(((const float*)&rb[i])[j]);
      *(v4bf*)&sB[buf][(bnq + j) * LDA + bkq] = t;
    }
  };

  v8f acc[2][4];
#pragma unroll
  for (int fm = 0; fm < 2; ++fm)
#pragma unroll
    for (int fn = 0; fn < 4; ++fn)
#pragma unroll
      for (int i = 0; i < 8; ++i) acc[fm][fn][i] = 0.0f;

  constexpr int KT = Hsz / BK;   // 16
  gload(0);
  lstore(0);
  __syncthreads();

  for (int t = 0; t < KT; ++t) {
    const int cur = t & 1;
    if (t + 1 < KT) gload((t + 1) * BK);

    v16bf af[2], bf_[4];
#pragma unroll
    for (int fm = 0; fm < 2; ++fm)
      af[fm] = load_frag_A(&sA[cur][(wm + fm * 16) * LDA], lane);
#pragma unroll
    for (int fn = 0; fn < 4; ++fn)
      bf_[fn] = load_frag_B(&sB[cur][(wn + fn * 16) * LDA], lane);

#pragma unroll
    for (int fm = 0; fm < 2; ++fm)
#pragma unroll
      for (int fn = 0; fn < 4; ++fn)
        acc[fm][fn] = __builtin_amdgcn_wmma_f32_16x16x32_bf16(
            false, af[fm], false, bf_[fn], (short)0, acc[fm][fn], false, false);

    if (t + 1 < KT) lstore(1 - cur);
    __syncthreads();
  }

  float bias[4];
#pragma unroll
  for (int fn = 0; fn < 4; ++fn)
    bias[fn] = b2[s * Psz + wn + fn * 16 + (lane & 15)];

  const int mro = (lane >> 4) * 8;
#pragma unroll
  for (int fm = 0; fm < 2; ++fm)
#pragma unroll
    for (int fn = 0; fn < 4; ++fn) {
      const int col = s * Psz + wn + fn * 16 + (lane & 15);
#pragma unroll
      for (int r = 0; r < 8; ++r) {
        const int row = m0 + wm + fm * 16 + mro + r;
        out[(size_t)row * (Ssz * Psz) + col] = acc[fm][fn][r] + bias[fn];
      }
    }
}

// =====================================================================
extern "C" void kernel_launch(void* const* d_in, const int* in_sizes, int n_in,
                              void* d_out, int out_size, void* d_ws, size_t ws_size,
                              hipStream_t stream) {
  (void)in_sizes; (void)n_in; (void)out_size; (void)ws_size;
  const float* x  = (const float*)d_in[0];
  const float* W1 = (const float*)d_in[1];
  const float* b1 = (const float*)d_in[2];
  const float* W2 = (const float*)d_in[3];
  const float* b2 = (const float*)d_in[4];
  float*  out = (float*)d_out;
  __bf16* hws = (__bf16*)d_ws;   // (B, S, H) bf16 = 37.75 MB

  dim3 blk(256);
  dim3 g1((Bsz / BM) * (Hsz / BN), Ssz);   // (32, 36): same-s tiles adjacent
  k_gemm1_gelu<<<g1, blk, 0, stream>>>(x, W1, b1, hws);

  dim3 g2(Bsz / BM, Ssz);                  // (8, 36)
  k_gemm2<<<g2, blk, 0, stream>>>(hws, W2, b2, out);
}
